// Seq2SeqLSTMModel_20160576487520
// MI455X (gfx1250) — matrix operand order repaired and checked by reading, not compiled, not run on hardware
//
#include <hip/hip_runtime.h>

typedef __attribute__((ext_vector_type(16))) _Float16 v16h;
typedef __attribute__((ext_vector_type(8)))  _Float16 v8h;
typedef __attribute__((ext_vector_type(8)))  float    v8f;

#define HDIM   128
#define GATES  512          // 4*H
#define T_HIST 50
#define T_OUT  60
#define NIN    6
#define NOUT   2
#define NAGENT 8
#define BATCH  16384

#define ROWS_PER_WG 16      // one 16-row batch tile per workgroup
#define WAVES_PER_WG 8      // wave w owns gate column tile jt=w
#define HS (ROWS_PER_WG * HDIM)

// combined-weight workspace layout (in halves): W[n][k], row-major, k-contiguous
#define KE0 160              // [Whh(128) | Wih(6) pad 32]
#define KL1 256              // [Whh(128) | Wih(128)]
#define OFF_E0 0
#define OFF_E1 (OFF_E0 + GATES * KE0)
#define OFF_D0 (OFF_E1 + GATES * KL1)
#define OFF_D1 (OFF_D0 + GATES * KE0)
#define W_TOTAL (OFF_D1 + GATES * KL1)   // 425984 halves

// ---------------- prep: build combined f16 weights in workspace ----------------
__global__ void prep_weights(const float* __restrict__ eWih0, const float* __restrict__ eWhh0,
                             const float* __restrict__ eWih1, const float* __restrict__ eWhh1,
                             const float* __restrict__ dWih0, const float* __restrict__ dWhh0,
                             const float* __restrict__ dWih1, const float* __restrict__ dWhh1,
                             _Float16* __restrict__ ws)
{
    int idx = blockIdx.x * blockDim.x + threadIdx.x;
    for (int i = idx; i < W_TOTAL; i += gridDim.x * blockDim.x) {
        float v;
        if (i < OFF_E1) {
            int r = i - OFF_E0, n = r / KE0, k = r % KE0;
            v = (k < HDIM) ? eWhh0[n * HDIM + k]
                           : ((k - HDIM) < NIN ? eWih0[n * NIN + (k - HDIM)] : 0.0f);
        } else if (i < OFF_D0) {
            int r = i - OFF_E1, n = r / KL1, k = r % KL1;
            v = (k < HDIM) ? eWhh1[n * HDIM + k] : eWih1[n * HDIM + (k - HDIM)];
        } else if (i < OFF_D1) {
            int r = i - OFF_D0, n = r / KE0, k = r % KE0;
            v = (k < HDIM) ? dWhh0[n * HDIM + k]
                           : ((k - HDIM) < NOUT ? dWih0[n * NOUT + (k - HDIM)] : 0.0f);
        } else {
            int r = i - OFF_D1, n = r / KL1, k = r % KL1;
            v = (k < HDIM) ? dWhh1[n * HDIM + k] : dWih1[n * HDIM + (k - HDIM)];
        }
        ws[i] = (_Float16)v;
    }
}

// ---------------- device helpers ----------------
// fast sigmoid / tanh: v_exp_f32 + v_rcp_f32 (no precise-division sequence)
__device__ __forceinline__ float sigmf(float x) {
    return __builtin_amdgcn_rcpf(1.0f + __expf(-x));
}
__device__ __forceinline__ float tanhfast(float x) {
    return 2.0f * __builtin_amdgcn_rcpf(1.0f + __expf(-2.0f * x)) - 1.0f;
}

__device__ __forceinline__ v16h load_afrag(const _Float16* p) {
    v8h lo = *(const v8h*)p;
    v8h hh = *(const v8h*)(p + 8);
    return __builtin_shufflevector(lo, hh, 0,1,2,3,4,5,6,7,8,9,10,11,12,13,14,15);
}

// One LSTM cell step; this wave computes gate-column tile jt (16 gate columns per
// gate group) for the WG's 16 batch rows.
// KB = K/32 blocks (5 or 8). XMODE: 0 = x from global history (IN=6),
// 1 = x from LDS half buffer (H=128, kb 4..7), 2 = x from LDS y buffer (OUT=2).
template<int KB, int XMODE>
__device__ __forceinline__ void lstm_cell(
    const _Float16* __restrict__ Wc,    // global combined weights [512][KB*32]
    const float*    __restrict__ biasL, // LDS bias [512]
    const _Float16* hcur,               // LDS [16][128] old h (shared, read-only here)
    _Float16*       hnew,               // LDS [16][128] new h (this wave writes cols jt*16..+15)
    float*          cbuf,               // LDS [16][128] c (in-place, disjoint cols per wave)
    const _Float16* xh,                 // LDS x buffer (XMODE==1)
    const float*    ybuf,               // LDS y buffer (XMODE==2)
    const float*    histx,              // global history base (XMODE==0)
    int lane, int jt)
{
    const int col = lane & 15;
    const int hi  = lane >> 4;
    const int Kc  = KB * 32;

    v8f acc[4];
    #pragma unroll
    for (int g = 0; g < 4; ++g) {
        float bv = biasL[g * HDIM + jt * 16 + col];
        #pragma unroll
        for (int r = 0; r < 8; ++r) acc[g][r] = bv;
    }

    const _Float16* ap  = hcur + col * HDIM + hi * 16;
    const _Float16* bp0 = Wc + (size_t)(jt * 16 + col) * Kc + hi * 16;

    // ---- h contribution: kb = 0..3 (rolled loop -> bounded register pressure) ----
    #pragma unroll 1
    for (int kb = 0; kb < 4; ++kb) {
        v16h a = load_afrag(ap + kb * 32);
        #pragma unroll
        for (int g = 0; g < 4; ++g) {
            v16h b = *(const v16h*)(bp0 + (size_t)g * HDIM * Kc + kb * 32);
            acc[g] = __builtin_amdgcn_wmma_f32_16x16x32_f16(
                         false, a, false, b, (short)0, acc[g], false, false);
        }
    }

    // ---- x contribution ----
    if (XMODE == 1) {
        const _Float16* xp = xh + col * HDIM + hi * 16;
        #pragma unroll 1
        for (int kb = 0; kb < 4; ++kb) {
            v16h a = load_afrag(xp + kb * 32);
            #pragma unroll
            for (int g = 0; g < 4; ++g) {
                v16h b = *(const v16h*)(bp0 + (size_t)g * HDIM * Kc + (4 + kb) * 32);
                acc[g] = __builtin_amdgcn_wmma_f32_16x16x32_f16(
                             false, a, false, b, (short)0, acc[g], false, false);
            }
        }
    } else {
        v16h a;
        #pragma unroll
        for (int i = 0; i < 16; ++i) a[i] = (_Float16)0.0f;
        if (XMODE == 0) {
            if (hi == 0) {
                #pragma unroll
                for (int i = 0; i < NIN; ++i)
                    a[i] = (_Float16)histx[col * (NAGENT * T_HIST * NIN) + i];
            }
        } else { // XMODE == 2
            if (hi == 0) {
                a[0] = (_Float16)ybuf[col * 2 + 0];
                a[1] = (_Float16)ybuf[col * 2 + 1];
            }
        }
        #pragma unroll
        for (int g = 0; g < 4; ++g) {
            v16h b = *(const v16h*)(bp0 + (size_t)g * HDIM * Kc + 4 * 32);
            acc[g] = __builtin_amdgcn_wmma_f32_16x16x32_f16(
                         false, a, false, b, (short)0, acc[g], false, false);
        }
    }

    // combine: D layout -> row m = r + 8*hi, col j = jt*16 + (lane&15)
    #pragma unroll
    for (int r = 0; r < 8; ++r) {
        int m = r + hi * 8;
        int j = jt * 16 + col;
        float cold = cbuf[m * HDIM + j];
        float ig = sigmf(acc[0][r]);
        float fg = sigmf(acc[1][r]);
        float gg = tanhfast(acc[2][r]);
        float og = sigmf(acc[3][r]);
        float cn = fg * cold + ig * gg;
        cbuf[m * HDIM + j] = cn;
        hnew[m * HDIM + j] = (_Float16)(og * tanhfast(cn));
    }
}

// ---------------- main fused seq2seq kernel ----------------
__global__ void __launch_bounds__(WAVES_PER_WG * 32)
__attribute__((amdgpu_waves_per_eu(4)))
lstm_seq2seq(const float* __restrict__ history, const float* __restrict__ start_token,
             const float* __restrict__ enc_b0,  const float* __restrict__ enc_b1,
             const float* __restrict__ dec_b0,  const float* __restrict__ dec_b1,
             const float* __restrict__ out_W,   const float* __restrict__ out_b,
             const _Float16* __restrict__ Wc,   float* __restrict__ out)
{
    __shared__ _Float16 Hb0[2][HS];   // layer0 h ping-pong (f16)
    __shared__ _Float16 Hb1[2][HS];   // layer1 h ping-pong (f16)
    __shared__ float    C0[HS];       // layer0 c
    __shared__ float    C1[HS];       // layer1 c
    __shared__ float    Y[ROWS_PER_WG * NOUT];
    __shared__ float    Bias[4][GATES];

    const int lane = threadIdx.x & 31;
    const int wave = threadIdx.x >> 5;   // == jt tile owned by this wave
    const int b0   = blockIdx.x * ROWS_PER_WG;

    // stage biases, zero-init state
    for (int i = threadIdx.x; i < GATES; i += WAVES_PER_WG * 32) {
        Bias[0][i] = enc_b0[i];
        Bias[1][i] = enc_b1[i];
        Bias[2][i] = dec_b0[i];
        Bias[3][i] = dec_b1[i];
    }
    for (int i = threadIdx.x; i < HS; i += WAVES_PER_WG * 32) {
        Hb0[0][i] = (_Float16)0.0f;
        Hb1[0][i] = (_Float16)0.0f;
        C0[i] = 0.0f;
        C1[i] = 0.0f;
    }
    __syncthreads();

    const _Float16* We0 = Wc + OFF_E0;
    const _Float16* We1 = Wc + OFF_E1;
    const _Float16* Wd0 = Wc + OFF_D0;
    const _Float16* Wd1 = Wc + OFF_D1;

    int cur0 = 0, cur1 = 0;

    // ---------------- encoder ----------------
    #pragma unroll 1
    for (int t = 0; t < T_HIST; ++t) {
        const float* histx = history + (size_t)b0 * (NAGENT * T_HIST * NIN) + (size_t)t * NIN;
        lstm_cell<5, 0>(We0, Bias[0], Hb0[cur0], Hb0[cur0 ^ 1], C0,
                        nullptr, nullptr, histx, lane, wave);
        cur0 ^= 1;
        __syncthreads();
        lstm_cell<8, 1>(We1, Bias[1], Hb1[cur1], Hb1[cur1 ^ 1], C1,
                        Hb0[cur0], nullptr, nullptr, lane, wave);
        cur1 ^= 1;
        __syncthreads();
    }

    // decoder input x0 = start_token broadcast
    if (wave == 0) Y[(lane & 15) * 2 + (lane >> 4)] = start_token[lane >> 4];
    __syncthreads();

    // ---------------- decoder ----------------
    #pragma unroll 1
    for (int t = 0; t < T_OUT; ++t) {
        lstm_cell<5, 2>(Wd0, Bias[2], Hb0[cur0], Hb0[cur0 ^ 1], C0,
                        nullptr, Y, nullptr, lane, wave);
        cur0 ^= 1;
        __syncthreads();
        lstm_cell<8, 1>(Wd1, Bias[3], Hb1[cur1], Hb1[cur1 ^ 1], C1,
                        Hb0[cur0], nullptr, nullptr, lane, wave);
        cur1 ^= 1;
        __syncthreads();

        // projection y = h1 @ out_W.T + out_b : lane -> (m = lane&15, o = lane>>4)
        if (wave == 0) {
            const _Float16* h1n = Hb1[cur1];
            int m = lane & 15, o = lane >> 4;
            float acc = out_b[o];
            #pragma unroll
            for (int jb = 0; jb < HDIM / 8; ++jb) {
                v8h hv = *(const v8h*)(h1n + m * HDIM + jb * 8);
                #pragma unroll
                for (int i = 0; i < 8; ++i)
                    acc += (float)hv[i] * out_W[o * HDIM + jb * 8 + i];
            }
            Y[m * 2 + o] = acc;
            out[((size_t)(b0 + m) * T_OUT + t) * NOUT + o] = acc;
        }
        __syncthreads();
    }
}

// ---------------- host launch ----------------
extern "C" void kernel_launch(void* const* d_in, const int* in_sizes, int n_in,
                              void* d_out, int out_size, void* d_ws, size_t ws_size,
                              hipStream_t stream) {
    const float* history     = (const float*)d_in[0];
    const float* start_token = (const float*)d_in[1];
    const float* enc_Wih0    = (const float*)d_in[2];
    const float* enc_Whh0    = (const float*)d_in[3];
    const float* enc_b0      = (const float*)d_in[4];
    const float* enc_Wih1    = (const float*)d_in[5];
    const float* enc_Whh1    = (const float*)d_in[6];
    const float* enc_b1      = (const float*)d_in[7];
    const float* dec_Wih0    = (const float*)d_in[8];
    const float* dec_Whh0    = (const float*)d_in[9];
    const float* dec_b0      = (const float*)d_in[10];
    const float* dec_Wih1    = (const float*)d_in[11];
    const float* dec_Whh1    = (const float*)d_in[12];
    const float* dec_b1      = (const float*)d_in[13];
    const float* out_W       = (const float*)d_in[14];
    const float* out_b       = (const float*)d_in[15];

    _Float16* Wc = (_Float16*)d_ws;

    prep_weights<<<(W_TOTAL + 255) / 256, 256, 0, stream>>>(
        enc_Wih0, enc_Whh0, enc_Wih1, enc_Whh1,
        dec_Wih0, dec_Whh0, dec_Wih1, dec_Whh1, Wc);

    lstm_seq2seq<<<BATCH / ROWS_PER_WG, WAVES_PER_WG * 32, 0, stream>>>(
        history, start_token, enc_b0, enc_b1, dec_b0, dec_b1,
        out_W, out_b, Wc, (float*)d_out);
}
